// SelfMatchingAttention_Loop_61710090108943
// MI455X (gfx1250) — compile-verified, matrix operand order checked by reading, not run on hardware
//
#include <hip/hip_runtime.h>
#include <hip/hip_bf16.h>

typedef __attribute__((ext_vector_type(16))) _Float16 v16h;
typedef __attribute__((ext_vector_type(8)))  float    v8f;

#define TT 256
#define BB 8
#define HH 128

// ---------------------------------------------------------------------------
// WMMA fragment loader (wave32, 16x16x32 f16 layout), f16 source (global or LDS).
// A (16x32): lanes 0-15 -> M=lane, halves 0-7 = K[kb..kb+7], 8-15 = K[kb+16..kb+23]
//            lanes 16-31 -> M=lane-16, halves 0-7 = K[kb+8..], 8-15 = K[kb+24..]
// B (32x16): same indexing with N in place of M (row-major W[N][K] == B[k][n]).
// Each lane: two contiguous 16-byte runs -> 2x b128 loads, no conversions.
// ---------------------------------------------------------------------------
__device__ __forceinline__ v16h load_frag_f16(const _Float16* __restrict__ base, int ld,
                                              int row_base, int kbase) {
  const int lane = threadIdx.x & 31;
  const _Float16* p = base + (size_t)(row_base + (lane & 15)) * ld + kbase + ((lane >> 4) << 3);
  v16h f;
#pragma unroll
  for (int j = 0; j < 8; ++j) {
    f[j]     = p[j];
    f[j + 8] = p[j + 16];
  }
  return f;
}

// ---------------------------------------------------------------------------
// One-time f32 -> f16 operand conversion (n multiple of 4)
// ---------------------------------------------------------------------------
__global__ __launch_bounds__(256) void cvt_f16(const float* __restrict__ in,
                                               _Float16* __restrict__ out, int n) {
  const int i = (blockIdx.x * 256 + threadIdx.x) * 4;
  if (i < n) {
    const float4 v = *(const float4*)(in + i);
    out[i]     = (_Float16)v.x;
    out[i + 1] = (_Float16)v.y;
    out[i + 2] = (_Float16)v.z;
    out[i + 3] = (_Float16)v.w;
  }
}

// ---------------------------------------------------------------------------
// GEMM (f16 operands, f32 accum): C[M,N] = X[M,K] * W[N,K]^T (+ bias[N]).
// One wave per 16x16 tile.
// ---------------------------------------------------------------------------
__global__ __launch_bounds__(256) void wmma_gemm16(const _Float16* __restrict__ X,
                                                   const _Float16* __restrict__ W,
                                                   const float* __restrict__ bias,
                                                   float* __restrict__ C,
                                                   int M, int N, int K) {
  const int wave   = threadIdx.x >> 5;
  const int tile   = blockIdx.x * 8 + wave;
  const int tilesN = N >> 4;
  const int nTiles = (M >> 4) * tilesN;
  if (tile >= nTiles) return;
  const int tm   = tile / tilesN;
  const int tn   = tile - tm * tilesN;
  const int lane = threadIdx.x & 31;

  v8f acc = {};
#pragma unroll 4
  for (int k = 0; k < K; k += 32) {
    v16h a = load_frag_f16(X, K, tm << 4, k);
    v16h b = load_frag_f16(W, K, tn << 4, k);
    acc = __builtin_amdgcn_wmma_f32_16x16x32_f16(false, a, false, b, (short)0, acc, false, false);
  }

  const int col = (tn << 4) + (lane & 15);
  const float bv = bias ? bias[col] : 0.f;
#pragma unroll
  for (int r = 0; r < 8; ++r) {
    const int row = (tm << 4) + ((lane >> 4) << 3) + r;
    C[(size_t)row * N + col] = acc[r] + bv;
  }
}

// ---------------------------------------------------------------------------
// Gate GEMM: UGh[M,N] = (f16)( U32[M,N] * sigmoid( Uh[M,K] * Wg[N,K]^T ) )
// ---------------------------------------------------------------------------
__global__ __launch_bounds__(256) void wmma_gemm_gate(const _Float16* __restrict__ Uh,
                                                      const _Float16* __restrict__ Wg,
                                                      const float* __restrict__ U32,
                                                      _Float16* __restrict__ UGh,
                                                      int M, int N, int K) {
  const int wave   = threadIdx.x >> 5;
  const int tile   = blockIdx.x * 8 + wave;
  const int tilesN = N >> 4;
  const int nTiles = (M >> 4) * tilesN;
  if (tile >= nTiles) return;
  const int tm   = tile / tilesN;
  const int tn   = tile - tm * tilesN;
  const int lane = threadIdx.x & 31;

  v8f acc = {};
#pragma unroll 4
  for (int k = 0; k < K; k += 32) {
    v16h a = load_frag_f16(Uh, K, tm << 4, k);
    v16h b = load_frag_f16(Wg, K, tn << 4, k);
    acc = __builtin_amdgcn_wmma_f32_16x16x32_f16(false, a, false, b, (short)0, acc, false, false);
  }

  const int col = (tn << 4) + (lane & 15);
#pragma unroll
  for (int r = 0; r < 8; ++r) {
    const int row = (tm << 4) + ((lane >> 4) << 3) + r;
    const float g = 1.f / (1.f + __expf(-acc[r]));
    UGh[(size_t)row * N + col] = (_Float16)(U32[(size_t)row * N + col] * g);
  }
}

// ---------------------------------------------------------------------------
// Attention: one wave per (q,b).  scores[k] = v . tanh(pbar[q]+prep[k]);
// softmax over k; ct = sum_k a[k]*pbar[k].
// Writes U32[q,b,:] = {pbar, ct} (f32, gate epilogue) and Uh (f16, GEMM A input).
// ---------------------------------------------------------------------------
__global__ __launch_bounds__(32) void attention_kernel(const float* __restrict__ prep,
                                                       const float* __restrict__ pbar,
                                                       const float* __restrict__ vT,
                                                       float* __restrict__ U32,
                                                       _Float16* __restrict__ Uh) {
  const int q = blockIdx.x;
  const int b = blockIdx.y;
  const int lane = threadIdx.x;
  __shared__ float sc[TT];

  float pb[4], vv[4];
  const float* pq = pbar + ((size_t)(q * BB + b)) * HH;
#pragma unroll
  for (int j = 0; j < 4; ++j) {
    pb[j] = pq[lane + 32 * j];
    vv[j] = vT[lane + 32 * j];
  }

  // pass 1: scores
  for (int k = 0; k < TT; ++k) {
    const float* pk = prep + ((size_t)(k * BB + b)) * HH;
    float s = 0.f;
#pragma unroll
    for (int j = 0; j < 4; ++j) s += vv[j] * tanhf(pb[j] + pk[lane + 32 * j]);
#pragma unroll
    for (int off = 16; off > 0; off >>= 1) s += __shfl_xor(s, off, 32);
    if (lane == 0) sc[k] = s;
  }
  __syncthreads();

  // softmax over k
  float m = -3.402823466e+38f;
  for (int k = lane; k < TT; k += 32) m = fmaxf(m, sc[k]);
#pragma unroll
  for (int off = 16; off > 0; off >>= 1) m = fmaxf(m, __shfl_xor(m, off, 32));
  float sum = 0.f;
  for (int k = lane; k < TT; k += 32) {
    const float e = __expf(sc[k] - m);
    sc[k] = e;
    sum += e;
  }
#pragma unroll
  for (int off = 16; off > 0; off >>= 1) sum += __shfl_xor(sum, off, 32);
  const float inv = 1.f / sum;
  __syncthreads();

  // pass 2: ct = sum_k a[k] * pbar[k,b,:]
  float ct[4] = {0.f, 0.f, 0.f, 0.f};
  for (int k = 0; k < TT; ++k) {
    const float a = sc[k];
    const float* pk = pbar + ((size_t)(k * BB + b)) * HH;
#pragma unroll
    for (int j = 0; j < 4; ++j) ct[j] += a * pk[lane + 32 * j];
  }

  float*     u  = U32 + ((size_t)(q * BB + b)) * (2 * HH);
  _Float16*  uh = Uh  + ((size_t)(q * BB + b)) * (2 * HH);
#pragma unroll
  for (int j = 0; j < 4; ++j) {
    const float c = ct[j] * inv;
    u[lane + 32 * j]       = pb[j];
    u[HH + lane + 32 * j]  = c;
    uh[lane + 32 * j]      = (_Float16)pb[j];
    uh[HH + lane + 32 * j] = (_Float16)c;
  }
}

// ---------------------------------------------------------------------------
// GRU scan (one layer; both cells scanned forward in time, matching the
// reference's jax.lax.scan).  Single workgroup, 256 threads = 8 waves.
//  - Whh (both cells) f16 in LDS (192 KB); hidden state kept in f32 + f16 LDS
//  - per step: gh[16,384] = h16[16,128] @ Whh^T via WMMA, K-outer/N-inner with
//    6 accumulators per wave (waves 0-3 = fwd cell, 4-7 = bwd cell)
//  - VALU gate update using precomputed GI = x@Wih^T + bih
//  - layer output written as f16 [T,B,2H] (next layer's GEMM input) or as
//    f32 [B,T,2H] directly into d_out for the last layer.
// ---------------------------------------------------------------------------
__global__ __launch_bounds__(256) void gru_scan(const float* __restrict__ GIf,
                                                const float* __restrict__ GIb,
                                                const float* __restrict__ Whhf,
                                                const float* __restrict__ Whhb,
                                                const float* __restrict__ bhhf,
                                                const float* __restrict__ bhhb,
                                                _Float16* __restrict__ outh,
                                                float* __restrict__ outf) {
  extern __shared__ char smem[];
  _Float16* whh  = (_Float16*)smem;                                    // [2][384*128]
  float*    hbuf = (float*)(smem + 2 * 384 * 128 * sizeof(_Float16));  // [2][16*128]
  float*    ghb  = hbuf + 2 * 16 * 128;                                // [2][8*384]
  _Float16* h16  = (_Float16*)(ghb + 2 * 8 * 384);                     // [2][16*128]

  const int tid = threadIdx.x;

  for (int i = tid; i < 2 * 384 * 128; i += 256) {
    const int d = i / (384 * 128);
    const int r = i - d * (384 * 128);
    whh[i] = (_Float16)(d ? Whhb[r] : Whhf[r]);
  }
  for (int i = tid; i < 2 * 16 * 128; i += 256) {
    hbuf[i] = 0.f;
    h16[i]  = (_Float16)0.f;
  }
  __syncthreads();

  const int lane = tid & 31;
  const int wave = tid >> 5;
  const int dir  = wave >> 2;                       // waves 0-3: fwd, 4-7: bwd
  const _Float16* whh_d = whh + dir * (384 * 128);
  const _Float16* h16_d = h16 + dir * (16 * 128);
  float*          gh_d  = ghb + dir * (8 * 384);
  const int       ntb   = (wave & 3) * 6;           // this wave's first N-tile

  for (int t = 0; t < TT; ++t) {
    // ---- recurrent matmul: gh = h @ Whh^T (A loaded once per K-step) ----
    v8f acc[6] = {};
#pragma unroll
    for (int kk = 0; kk < 4; ++kk) {
      v16h a = load_frag_f16(h16_d, 128, 0, kk * 32);
#pragma unroll
      for (int i = 0; i < 6; ++i) {
        v16h b = load_frag_f16(whh_d, 128, (ntb + i) << 4, kk * 32);
        acc[i] = __builtin_amdgcn_wmma_f32_16x16x32_f16(false, a, false, b, (short)0,
                                                        acc[i], false, false);
      }
    }
    if (lane < 16) {                                // rows 0-7 are real batch rows
#pragma unroll
      for (int i = 0; i < 6; ++i)
#pragma unroll
        for (int r = 0; r < 8; ++r) gh_d[r * 384 + ((ntb + i) << 4) + lane] = acc[i][r];
    }
    __syncthreads();

    // ---- gate update: 2 cells * 8 batch * 128 hidden = 2048 items ----
#pragma unroll
    for (int rep = 0; rep < 8; ++rep) {
      const int idx = tid + rep * 256;
      const int d2  = idx >> 10;
      const int rem = idx & 1023;
      const int b   = rem >> 7;
      const int j   = rem & 127;
      const float* gi  = (d2 ? GIb : GIf) + ((size_t)(t * BB + b)) * 384;
      const float* bhh = d2 ? bhhb : bhhf;
      const float* gh2 = ghb + d2 * (8 * 384) + b * 384;
      const float rr = 1.f / (1.f + __expf(-(gi[j] + gh2[j] + bhh[j])));
      const float zz = 1.f / (1.f + __expf(-(gi[128 + j] + gh2[128 + j] + bhh[128 + j])));
      const float nn = tanhf(gi[256 + j] + rr * (gh2[256 + j] + bhh[256 + j]));
      const int   hi = d2 * (16 * 128) + b * 128 + j;
      const float hnew = (1.f - zz) * nn + zz * hbuf[hi];
      hbuf[hi] = hnew;
      h16[hi]  = (_Float16)hnew;
      if (outh) outh[((size_t)t * BB + b) * (2 * HH) + d2 * HH + j] = (_Float16)hnew;
      if (outf) outf[((size_t)b * TT + t) * (2 * HH) + d2 * HH + j] = hnew;  // [B,T,2H]
    }
    __syncthreads();
  }
}

// ---------------------------------------------------------------------------
extern "C" void kernel_launch(void* const* d_in, const int* in_sizes, int n_in,
                              void* d_out, int out_size, void* d_ws, size_t ws_size,
                              hipStream_t stream) {
  (void)in_sizes; (void)n_in; (void)out_size; (void)ws_size;

  const float* passage = (const float*)d_in[0];   // [T,B,256]
  const float* vT      = (const float*)d_in[2];   // [1,128]
  const float* WvP     = (const float*)d_in[3];   // [128,256]
  const float* WvPbar  = (const float*)d_in[4];   // [128,256]
  const float* gate_w  = (const float*)d_in[5];   // [256,256]
  const float* Wih     = (const float*)d_in[6];   // [3,2,384,256]
  const float* Whh     = (const float*)d_in[7];   // [3,2,384,128]
  const float* bih     = (const float*)d_in[8];   // [3,2,384]
  const float* bhh     = (const float*)d_in[9];   // [3,2,384]
  float* out = (float*)d_out;                     // [B,T,256]

  // ---- workspace layout ----
  char* w = (char*)d_ws;
  float* prep = (float*)w;               w += (size_t)2048 * 128 * 4;
  float* pbar = (float*)w;               w += (size_t)2048 * 128 * 4;
  float* U32  = (float*)w;               w += (size_t)2048 * 256 * 4;
  float* GIf  = (float*)w;               w += (size_t)2048 * 384 * 4;
  float* GIb  = (float*)w;               w += (size_t)2048 * 384 * 4;
  _Float16* X16     = (_Float16*)w;      w += (size_t)2048 * 256 * 2;
  _Float16* WvP16   = (_Float16*)w;      w += (size_t)128 * 256 * 2;
  _Float16* WvPb16  = (_Float16*)w;      w += (size_t)128 * 256 * 2;
  _Float16* gate16  = (_Float16*)w;      w += (size_t)256 * 256 * 2;
  _Float16* Wih16   = (_Float16*)w;      w += (size_t)6 * 384 * 256 * 2;
  _Float16* Uh      = (_Float16*)w;      w += (size_t)2048 * 256 * 2;
  _Float16* UGh     = (_Float16*)w;      w += (size_t)2048 * 256 * 2;
  _Float16* seqA16  = (_Float16*)w;      w += (size_t)2048 * 256 * 2;
  _Float16* seqB16  = (_Float16*)w;      w += (size_t)2048 * 256 * 2;

  const dim3 blk(256);
  auto cvtgrid = [](int n) { return (n / 4 + 255) / 256; };

  // ---- one-time operand conversion to f16 ----
  cvt_f16<<<cvtgrid(2048 * 256), blk, 0, stream>>>(passage, X16, 2048 * 256);
  cvt_f16<<<cvtgrid(128 * 256), blk, 0, stream>>>(WvP, WvP16, 128 * 256);
  cvt_f16<<<cvtgrid(128 * 256), blk, 0, stream>>>(WvPbar, WvPb16, 128 * 256);
  cvt_f16<<<cvtgrid(256 * 256), blk, 0, stream>>>(gate_w, gate16, 256 * 256);
  cvt_f16<<<cvtgrid(6 * 384 * 256), blk, 0, stream>>>(Wih, Wih16, 6 * 384 * 256);

  // ---- projections: prep/pbar = X @ Wv^T ----
  wmma_gemm16<<<(2048 / 16) * (128 / 16) / 8, blk, 0, stream>>>(X16, WvP16, nullptr, prep,
                                                                2048, 128, 256);
  wmma_gemm16<<<(2048 / 16) * (128 / 16) / 8, blk, 0, stream>>>(X16, WvPb16, nullptr, pbar,
                                                                2048, 128, 256);

  // ---- attention + concat -> U (f32 + f16) ----
  attention_kernel<<<dim3(TT, BB), 32, 0, stream>>>(prep, pbar, vT, U32, Uh);

  // ---- gate: UGh = (f16)(U * sigmoid(U @ gate_w^T)) ----
  wmma_gemm_gate<<<(2048 / 16) * (256 / 16) / 8, blk, 0, stream>>>(Uh, gate16, U32, UGh,
                                                                   2048, 256, 256);

  const size_t smem = (size_t)2 * 384 * 128 * sizeof(_Float16)   // Whh f16
                    + (size_t)2 * 16 * 128 * sizeof(float)       // h f32
                    + (size_t)2 * 8 * 384 * sizeof(float)        // gh scratch
                    + (size_t)2 * 16 * 128 * sizeof(_Float16);   // h f16

  const _Float16* layer_in = UGh;
  for (int l = 0; l < 3; ++l) {
    const _Float16* Wih_f = Wih16 + (size_t)(l * 2 + 0) * 384 * 256;
    const _Float16* Wih_b = Wih16 + (size_t)(l * 2 + 1) * 384 * 256;
    const float* Whh_f = Whh + (size_t)(l * 2 + 0) * 384 * 128;
    const float* Whh_b = Whh + (size_t)(l * 2 + 1) * 384 * 128;
    const float* bih_f = bih + (size_t)(l * 2 + 0) * 384;
    const float* bih_b = bih + (size_t)(l * 2 + 1) * 384;
    const float* bhh_f = bhh + (size_t)(l * 2 + 0) * 384;
    const float* bhh_b = bhh + (size_t)(l * 2 + 1) * 384;

    // time-parallel input projections: GI = layer_in @ Wih^T + bih
    wmma_gemm16<<<(2048 / 16) * (384 / 16) / 8, blk, 0, stream>>>(layer_in, Wih_f, bih_f,
                                                                  GIf, 2048, 384, 256);
    wmma_gemm16<<<(2048 / 16) * (384 / 16) / 8, blk, 0, stream>>>(layer_in, Wih_b, bih_b,
                                                                  GIb, 2048, 384, 256);

    if (l < 2) {
      _Float16* lo = (l == 0) ? seqA16 : seqB16;
      gru_scan<<<1, blk, smem, stream>>>(GIf, GIb, Whh_f, Whh_b, bhh_f, bhh_b, lo, nullptr);
      layer_in = lo;
    } else {
      gru_scan<<<1, blk, smem, stream>>>(GIf, GIb, Whh_f, Whh_b, bhh_f, bhh_b, nullptr, out);
    }
  }
}
